// GPT_43361989820641
// MI455X (gfx1250) — matrix operand order repaired and checked by reading, not compiled, not run on hardware
//
#include <hip/hip_runtime.h>
#include <hip/hip_bf16.h>
#include <math.h>

// ---------------- model constants (match reference) ----------------
static constexpr int BB  = 2;      // batch
static constexpr int TT  = 1024;   // query length
static constexpr int SS  = 512;    // frames length
static constexpr int CC  = 1024;   // channels
static constexpr int HH  = 16;     // heads
static constexpr int HD  = 64;     // head dim
static constexpr int FF  = 4096;   // mlp hidden
static constexpr int LL  = 4;      // layers per stack
static constexpr int MQ  = BB * TT;   // 2048 query rows
static constexpr int MKV = BB * SS;   // 1024 frame rows

typedef unsigned short B16;
typedef __attribute__((ext_vector_type(16))) __bf16 v16bf;
typedef __attribute__((ext_vector_type(8)))  float  v8f;
typedef __attribute__((ext_vector_type(4)))  unsigned int v4u;
typedef __attribute__((ext_vector_type(8)))  int v8i;
typedef __attribute__((ext_vector_type(4)))  int v4i;

// Tensor Data Mover availability (device pass only). 5-arg = ROCm7.2 clang-22,
// 6-arg = therock/clang-23 (ships the gfx1250 TDM header).
#if defined(__HIP_DEVICE_COMPILE__) && __has_builtin(__builtin_amdgcn_tensor_load_to_lds)
  #define HAVE_TDM 1
  #if __has_include(<hip/amd_detail/amd_gfx1250_TDM.h>)
    #define TDM_6ARG 1
  #else
    #define TDM_6ARG 0
  #endif
#else
  #define HAVE_TDM 0
#endif

union BFrag { v16bf v; B16 u[16]; uint4 q[2]; };

__device__ __forceinline__ B16 f2bf(float x) {
  unsigned u = __float_as_uint(x);
  unsigned r = (u + 0x7FFFu + ((u >> 16) & 1u)) >> 16;
  return (B16)r;
}
__device__ __forceinline__ float bf2f(B16 h) {
  return __uint_as_float(((unsigned)h) << 16);
}
__device__ __forceinline__ v8f bwmma(const BFrag& a, const BFrag& b, v8f c) {
  return __builtin_amdgcn_wmma_f32_16x16x32_bf16(false, a.v, false, b.v,
                                                 (short)0, c, false, false);
}
__device__ __forceinline__ float wredmax(float v){ for(int m=16;m>0;m>>=1) v=fmaxf(v,__shfl_xor(v,m,32)); return v; }
__device__ __forceinline__ float wredmin(float v){ for(int m=16;m>0;m>>=1) v=fminf(v,__shfl_xor(v,m,32)); return v; }
__device__ __forceinline__ float wredsum(float v){ for(int m=16;m>0;m>>=1) v+=__shfl_xor(v,m,32); return v; }
__device__ __forceinline__ int   wredsumi(int v){ for(int m=16;m>0;m>>=1) v+=__shfl_xor(v,m,32); return v; }

#if HAVE_TDM
// Issue a 2D TDM load: tileH x tileW bf16 elements from a row-major tensor with
// row stride rowStride (elements) into LDS at ldsAddr (bytes). D# packing per
// CDNA5 ISA (group0: count/lds/global/type,
// group1: data_size + dims/strides).
__device__ __forceinline__ void tdm_load_tile(unsigned ldsAddr, const B16* gptr,
                                              unsigned tileW, unsigned tileH,
                                              unsigned rowStride,
                                              unsigned tensorW, unsigned tensorH) {
  unsigned long long ga = (unsigned long long)gptr;      // byte address
  v4u g0;
  g0.x = 1u;                                             // count=1, user mode
  g0.y = ldsAddr;                                        // lds_addr [63:32]
  g0.z = (unsigned)(ga & 0xFFFFFFFFu);                   // global_addr[31:0]
  g0.w = (unsigned)((ga >> 32) & 0x1FFFFFFu) | (2u << 30); // addr[56:32] | type=2
  v8i g1;
  g1[0] = (int)(1u << 16);                               // data_size=1 -> 2 bytes
  g1[1] = (int)((tensorW & 0xFFFFu) << 16);              // tensor_dim0[15:0]
  g1[2] = (int)(((tensorW >> 16) & 0xFFFFu) |            // tensor_dim0[31:16]
                ((tensorH & 0xFFFFu) << 16));            // tensor_dim1[15:0]
  g1[3] = (int)(((tensorH >> 16) & 0xFFFFu) |            // tensor_dim1[31:16]
                ((tileW & 0xFFFFu) << 16));              // tile_dim0
  g1[4] = (int)(tileH & 0xFFFFu);                        // tile_dim1 (tile_dim2=0)
  g1[5] = (int)rowStride;                                // tensor_dim0_stride[31:0]
  g1[6] = 0;                                             // stride[47:32], dim1_stride lo
  g1[7] = 0;
  v4i z4 = {0, 0, 0, 0};
#if TDM_6ARG
  v8i z8 = {0, 0, 0, 0, 0, 0, 0, 0};
  __builtin_amdgcn_tensor_load_to_lds(g0, g1, z4, z4, z8, 0);
#else
  __builtin_amdgcn_tensor_load_to_lds(g0, g1, z4, z4, 0);
#endif
}
#endif

// ---------------- elementwise helpers ----------------
__global__ void k_cast_bf16(const float* __restrict__ in, B16* __restrict__ out, int n) {
  int i = blockIdx.x * blockDim.x + threadIdx.x;
  if (i < n) out[i] = f2bf(in[i]);
}
__global__ void k_copy_cast(const float* __restrict__ in, float* __restrict__ outF,
                            B16* __restrict__ outB, int n) {
  int i = blockIdx.x * blockDim.x + threadIdx.x;
  if (i < n) { float v = in[i]; outF[i] = v; outB[i] = f2bf(v); }
}
__global__ void k_add_inplace(float* __restrict__ a, const float* __restrict__ b, int n) {
  int i = blockIdx.x * blockDim.x + threadIdx.x;
  if (i < n) a[i] += b[i];
}

// ---------------- layernorm: one 256-thread block per row ----------------
__global__ __launch_bounds__(256) void k_ln(const float* __restrict__ in,
                                            const float* __restrict__ g,
                                            const float* __restrict__ bt,
                                            float* __restrict__ outF,
                                            B16* __restrict__ outB) {
  const int row = blockIdx.x, tid = threadIdx.x;
  const float* x = in + (size_t)row * CC;
  float s1 = 0.f, s2 = 0.f;
  for (int i = tid; i < CC; i += 256) { float v = x[i]; s1 += v; s2 += v * v; }
  __shared__ float sh1[8], sh2[8];
  s1 = wredsum(s1); s2 = wredsum(s2);
  if ((tid & 31) == 0) { sh1[tid >> 5] = s1; sh2[tid >> 5] = s2; }
  __syncthreads();
  float t1 = 0.f, t2 = 0.f;
  #pragma unroll
  for (int i = 0; i < 8; ++i) { t1 += sh1[i]; t2 += sh2[i]; }
  const float mean = t1 / CC;
  const float var  = t2 / CC - mean * mean;
  const float rstd = rsqrtf(var + 1e-5f);
  for (int i = tid; i < CC; i += 256) {
    float v = (x[i] - mean) * rstd * g[i] + bt[i];
    size_t o = (size_t)row * CC + i;
    if (outF) outF[o] = v;
    if (outB) outB[o] = f2bf(v);
  }
}

// ---------------- v row-normalize per 64-elem head segment ----------------
__global__ __launch_bounds__(32) void k_vnorm(const float* __restrict__ vf,
                                              B16* __restrict__ vb) {
  const size_t base = (size_t)blockIdx.x * HD;
  const int lane = threadIdx.x;
  float v0 = vf[base + lane], v1 = vf[base + 32 + lane];
  float ss = wredsum(v0 * v0 + v1 * v1);
  float inv = 1.f / fmaxf(sqrtf(ss), 1e-12f);
  vb[base + lane]      = f2bf(v0 * inv);
  vb[base + 32 + lane] = f2bf(v1 * inv);
}

// ---------------- bf16 WMMA GEMM: O[M,N] = A[M,K] @ W[K,N] (+bias,+gelu,+resid) ----
// 256 threads = 8 waves, tile 64x128, K-step 32.
// A tile: TDM DMA (tensor_load_to_lds) double-buffered, wave0-issued,
//         synchronized with s_wait_tensorcnt + workgroup barrier.
// W tile: staged transposed [n][k] with packed 32-bit DS stores so B-fragments
//         are two contiguous 16B LDS loads per the ISA 16-bit B layout.
__global__ __launch_bounds__(256) void k_gemm(const B16* __restrict__ A,
                                              const B16* __restrict__ W,
                                              const float* __restrict__ bias,
                                              const float* __restrict__ resid,
                                              float* __restrict__ outF,
                                              B16* __restrict__ outB,
                                              int M, int N, int K, int gelu) {
  __shared__ B16 sA[2][64 * 32];
  __shared__ B16 sW[128 * 32];   // [n][k]
  const int tid  = threadIdx.x;
  const int lane = tid & 31, wave = tid >> 5;
  const int mi = wave >> 1, nq = wave & 1;
  const int rowBase = blockIdx.y * 64;
  const int colBase = blockIdx.x * 128;
  v8f acc[4] = {};
#if HAVE_TDM
  if (wave == 0)  // prologue: DMA first A tile (64x32) into buffer 0
    tdm_load_tile((unsigned)(unsigned long long)&sA[0][0],
                  A + (size_t)rowBase * K, 32, 64, (unsigned)K,
                  (unsigned)K, (unsigned)M);
#endif
  int it = 0;
  for (int k0 = 0; k0 < K; k0 += 32, ++it) {
    const int cur = it & 1;
#if HAVE_TDM
    if (wave == 0 && k0 + 32 < K)  // DMA next tile into the other buffer
      tdm_load_tile((unsigned)(unsigned long long)&sA[cur ^ 1][0],
                    A + (size_t)rowBase * K + (k0 + 32), 32, 64, (unsigned)K,
                    (unsigned)K, (unsigned)M);
#else
    {  // synchronous A staging fallback (one uint4 per thread)
      int r = tid >> 2, c = (tid & 3) * 8;
      *(uint4*)(&sA[cur][r * 32 + c]) =
          *(const uint4*)(A + (size_t)(rowBase + r) * K + k0 + c);
    }
#endif
    {  // stage W 32x128 transposed: 2 k-rows x 8 n per thread, packed b32 stores
      int k2 = (tid & 15) * 2, n0 = (tid >> 4) * 8;
      const B16* s0 = W + (size_t)(k0 + k2) * N + colBase + n0;
      const B16* s1 = s0 + N;
      B16 r0[8], r1[8];
      *(uint4*)r0 = *(const uint4*)s0;
      *(uint4*)r1 = *(const uint4*)s1;
      #pragma unroll
      for (int j = 0; j < 8; ++j) {
        unsigned pk = (unsigned)r0[j] | ((unsigned)r1[j] << 16);
        *(unsigned*)(&sW[(n0 + j) * 32 + k2]) = pk;
      }
      if (k0 + 32 < K)  // prefetch next W rows (global_prefetch_b8)
        __builtin_prefetch(W + (size_t)(k0 + 32 + k2) * N + colBase + n0, 0, 3);
    }
#if HAVE_TDM
    if (wave == 0) {   // current tile must have landed; next may stay in flight
      if (k0 + 32 < K) __builtin_amdgcn_s_wait_tensorcnt(1);
      else             __builtin_amdgcn_s_wait_tensorcnt(0);
    }
#endif
    __syncthreads();
    BFrag a;
    {
      int m = mi * 16 + (lane & 15);
      int kh = (lane >> 4) * 8;
      a.q[0] = *(const uint4*)(&sA[cur][m * 32 + kh]);
      a.q[1] = *(const uint4*)(&sA[cur][m * 32 + 16 + kh]);
    }
    #pragma unroll
    for (int j = 0; j < 4; ++j) {
      BFrag b;
      int n = nq * 64 + j * 16 + (lane & 15);
      int kh = (lane >> 4) * 8;
      b.q[0] = *(const uint4*)(&sW[n * 32 + kh]);
      b.q[1] = *(const uint4*)(&sW[n * 32 + kh + 16]);
      acc[j] = bwmma(a, b, acc[j]);
    }
    __syncthreads();
  }
  #pragma unroll
  for (int j = 0; j < 4; ++j) {
    int col = colBase + nq * 64 + j * 16 + (lane & 15);
    float bsv = bias ? bias[col] : 0.f;
    #pragma unroll
    for (int r = 0; r < 8; ++r) {
      int row = rowBase + mi * 16 + (lane >> 4) * 8 + r;
      float v = acc[j][r] + bsv;
      if (gelu) v = 0.5f * v * (1.f + erff(v * 0.70710678118654752f));
      size_t idx = (size_t)row * N + col;
      if (resid) v += resid[idx];
      if (outF) outF[idx] = v;
      if (outB) outB[idx] = f2bf(v);
    }
  }
}

// ---------------- sparse top-64 attention -----------------------------
// one wave per (16-query tile, head, batch): QK^T (WMMA) -> LDS scores ->
// bisection top-k threshold -> softmax in place -> PV (WMMA) -> y bf16.
__global__ __launch_bounds__(32) void k_attn(const B16* __restrict__ Q,
                                             const B16* __restrict__ Kx,
                                             const B16* __restrict__ V,
                                             B16* __restrict__ Y,
                                             int Skv, int causal) {
  __shared__ B16 sS[16 * 1024];
  __shared__ B16 sQ[16 * 64];
  const int lane = threadIdx.x;
  const int t0 = blockIdx.x * 16, h = blockIdx.y, b = blockIdx.z;
  const size_t qbase = ((size_t)(b * TT + t0)) * CC + h * HD;
  // stage q tile (16 x 64 halves)
  #pragma unroll
  for (int j = 0; j < 4; ++j) {
    int i = lane * 32 + j * 8;
    int m = i >> 6, c = i & 63;
    *(uint4*)(sQ + m * 64 + c) = *(const uint4*)(Q + qbase + (size_t)m * CC + c);
  }
  __syncthreads();
  BFrag a0, a1;
  {
    int m = lane & 15, kh = (lane >> 4) * 8;
    a0.q[0] = *(const uint4*)(sQ + m * 64 + kh);
    a0.q[1] = *(const uint4*)(sQ + m * 64 + 16 + kh);
    a1.q[0] = *(const uint4*)(sQ + m * 64 + 32 + kh);
    a1.q[1] = *(const uint4*)(sQ + m * 64 + 48 + kh);
  }
  // ---- scores = q @ k^T * 1/sqrt(hd), causal mask as -1e30 sentinel ----
  for (int s0 = 0; s0 < Skv; s0 += 16) {
    const int key = s0 + (lane & 15);
    const B16* krow = Kx + ((size_t)(b * Skv + key)) * CC + h * HD;
    const int kh = (lane >> 4) * 8;
    BFrag b0, b1;
    b0.q[0] = *(const uint4*)(krow + kh);
    b0.q[1] = *(const uint4*)(krow + kh + 16);
    b1.q[0] = *(const uint4*)(krow + 32 + kh);
    b1.q[1] = *(const uint4*)(krow + 32 + kh + 16);
    v8f acc = {};
    acc = bwmma(a0, b0, acc);
    acc = bwmma(a1, b1, acc);
    #pragma unroll
    for (int r = 0; r < 8; ++r) {
      int m = (lane >> 4) * 8 + r;
      float v = acc[r] * 0.125f;                     // 1/sqrt(64)
      if (causal && key > t0 + m) v = -1e30f;
      sS[m * 1024 + key] = f2bf(v);
    }
  }
  __syncthreads();
  // ---- per-row top-64 threshold (bisection) + softmax in place ----
  for (int m = 0; m < 16; ++m) {
    B16* row = sS + m * 1024;
    float vmax = -3e38f, vmin = 3e38f; int nvalid = 0;
    for (int i = lane; i < Skv; i += 32) {
      float v = bf2f(row[i]);
      if (v > -1e29f) { vmax = fmaxf(vmax, v); vmin = fminf(vmin, v); nvalid++; }
    }
    vmax = wredmax(vmax); vmin = wredmin(vmin); nvalid = wredsumi(nvalid);
    float thr = -3e38f;
    if (nvalid > 64) {
      float lo = vmin, hi = vmax;
      for (int it = 0; it < 24; ++it) {
        float mid = 0.5f * (lo + hi);
        int cnt = 0;
        for (int i = lane; i < Skv; i += 32) cnt += (bf2f(row[i]) >= mid);
        cnt = wredsumi(cnt);
        if (cnt >= 64) lo = mid; else hi = mid;
      }
      thr = lo;
    }
    float ssum = 0.f;
    for (int i = lane; i < Skv; i += 32) {
      float v = bf2f(row[i]);
      ssum += (v >= thr) ? __expf(v - vmax) : 0.f;
    }
    ssum = wredsum(ssum);
    const float inv = 1.f / ssum;
    for (int i = lane; i < Skv; i += 32) {
      float v = bf2f(row[i]);
      row[i] = f2bf((v >= thr) ? __expf(v - vmax) * inv : 0.f);
    }
  }
  __syncthreads();
  // ---- y = probs @ v ----
  v8f y[4] = {};
  for (int s0 = 0; s0 < Skv; s0 += 32) {
    BFrag a;
    {
      int m = lane & 15, kh = (lane >> 4) * 8;
      a.q[0] = *(const uint4*)(sS + m * 1024 + s0 + kh);
      a.q[1] = *(const uint4*)(sS + m * 1024 + s0 + 16 + kh);
    }
    #pragma unroll
    for (int f = 0; f < 4; ++f) {
      BFrag bb;
      #pragma unroll
      for (int i = 0; i < 16; ++i) {
        int key = s0 + (lane >> 4) * 8 + (i & 7) + (i >> 3) * 16;
        bb.u[i] = V[((size_t)(b * Skv + key)) * CC + h * HD + f * 16 + (lane & 15)];
      }
      y[f] = bwmma(a, bb, y[f]);
    }
  }
  #pragma unroll
  for (int f = 0; f < 4; ++f)
    #pragma unroll
    for (int r = 0; r < 8; ++r) {
      int t = t0 + (lane >> 4) * 8 + r;
      Y[((size_t)(b * TT + t)) * CC + h * HD + f * 16 + (lane & 15)] = f2bf(y[f][r]);
    }
}

// ---------------- host orchestration ----------------
extern "C" void kernel_launch(void* const* d_in, const int* in_sizes, int n_in,
                              void* d_out, int out_size, void* d_ws, size_t ws_size,
                              hipStream_t stream) {
  (void)in_sizes; (void)n_in; (void)out_size; (void)ws_size;
  const float* ids    = (const float*)d_in[0];
  const float* frames = (const float*)d_in[1];
  const float* LN[2]  = {(const float*)d_in[2],  (const float*)d_in[9]};
  const float* AW[2]  = {(const float*)d_in[3],  (const float*)d_in[10]};
  const float* AB[2]  = {(const float*)d_in[4],  (const float*)d_in[11]};
  const float* W1[2]  = {(const float*)d_in[5],  (const float*)d_in[12]};
  const float* B1[2]  = {(const float*)d_in[6],  (const float*)d_in[13]};
  const float* W2[2]  = {(const float*)d_in[7],  (const float*)d_in[14]};
  const float* B2[2]  = {(const float*)d_in[8],  (const float*)d_in[15]};
  const float* lnf_g  = (const float*)d_in[16];
  const float* lnf_b  = (const float*)d_in[17];
  float* out = (float*)d_out;

  char* ws = (char*)d_ws;
  size_t off = 0;
  auto alloc = [&](size_t bytes) -> char* {
    off = (off + 255) & ~(size_t)255;
    char* p = ws + off; off += bytes; return p;
  };
  float* xf   = (float*)alloc((size_t)MQ * CC * 4);
  B16*   xb   = (B16*)  alloc((size_t)MQ * CC * 2);
  float* hf   = (float*)alloc((size_t)MQ * CC * 4);
  float* h2f  = (float*)alloc((size_t)MQ * CC * 4);
  float* blf  = (float*)alloc((size_t)MQ * CC * 4);
  B16*   t0b  = (B16*)  alloc((size_t)MQ * CC * 2);
  B16*   qb   = (B16*)  alloc((size_t)MQ * CC * 2);
  B16*   kb   = (B16*)  alloc((size_t)MQ * CC * 2);
  float* vf   = (float*)alloc((size_t)MQ * CC * 4);
  B16*   vb   = (B16*)  alloc((size_t)MQ * CC * 2);
  B16*   yb   = (B16*)  alloc((size_t)MQ * CC * 2);
  B16*   m1b  = (B16*)  alloc((size_t)MQ * FF * 2);
  B16*   frb  = (B16*)  alloc((size_t)MKV * CC * 2);
  B16*   wbq  = (B16*)  alloc((size_t)CC * CC * 2);
  B16*   wbk  = (B16*)  alloc((size_t)CC * CC * 2);
  B16*   wbv  = (B16*)  alloc((size_t)CC * CC * 2);
  B16*   wbo  = (B16*)  alloc((size_t)CC * CC * 2);
  B16*   wb1  = (B16*)  alloc((size_t)CC * FF * 2);
  B16*   wb2  = (B16*)  alloc((size_t)FF * CC * 2);

  auto cast = [&](const float* src, B16* dst, int n) {
    k_cast_bf16<<<(n + 255) / 256, 256, 0, stream>>>(src, dst, n);
  };
  auto gemm = [&](const B16* A, const B16* W, const float* bias, const float* resid,
                  float* oF, B16* oB, int M, int N, int K, int gelu) {
    dim3 grid(N / 128, M / 64);
    k_gemm<<<grid, 256, 0, stream>>>(A, W, bias, resid, oF, oB, M, N, K, gelu);
  };
  auto lnorm = [&](const float* in, const float* g, const float* bt,
                   float* oF, B16* oB, int rows) {
    k_ln<<<rows, 256, 0, stream>>>(in, g, bt, oF, oB);
  };

  // x = ids (f32 working copy + bf16 copy for raw-kv use); frames -> bf16
  k_copy_cast<<<(MQ * CC + 255) / 256, 256, 0, stream>>>(ids, xf, xb, MQ * CC);
  cast(frames, frb, MKV * CC);

  for (int stack = 0; stack < 2; ++stack) {
    const int selfAttn = (stack == 0);
    const int Skv = selfAttn ? TT : SS;
    const int Mkv = selfAttn ? MQ : MKV;
    const B16* kvb = selfAttn ? xb : frb;
    for (int i = 0; i < LL; ++i) {
      const float* ln = LN[stack] + (size_t)i * 6 * CC;
      const float* aw = AW[stack] + (size_t)i * 4 * CC * CC;
      const float* ab = AB[stack] + (size_t)i * 4 * CC;
      const float* w1 = W1[stack] + (size_t)i * CC * FF;
      const float* b1 = B1[stack] + (size_t)i * FF;
      const float* w2 = W2[stack] + (size_t)i * FF * CC;
      const float* b2 = B2[stack] + (size_t)i * CC;
      // weights -> bf16 (per-layer reuse of the same buffers)
      cast(aw + 0 * CC * CC, wbq, CC * CC);
      cast(aw + 1 * CC * CC, wbk, CC * CC);
      cast(aw + 2 * CC * CC, wbv, CC * CC);
      cast(aw + 3 * CC * CC, wbo, CC * CC);
      cast(w1, wb1, CC * FF);
      cast(w2, wb2, FF * CC);
      // attn: h = x + proj(attn(ln(x), kv_raw))
      lnorm(xf, ln + 0 * CC, ln + 1 * CC, nullptr, t0b, MQ);
      gemm(t0b, wbq, ab + 0 * CC, nullptr, nullptr, qb, MQ,  CC, CC, 0);
      gemm(kvb, wbk, ab + 1 * CC, nullptr, nullptr, kb, Mkv, CC, CC, 0);
      gemm(kvb, wbv, ab + 2 * CC, nullptr, vf,      nullptr, Mkv, CC, CC, 0);
      k_vnorm<<<Mkv * HH, 32, 0, stream>>>(vf, vb);
      {
        dim3 grid(TT / 16, HH, BB);
        k_attn<<<grid, 32, 0, stream>>>(qb, kb, vb, yb, Skv, selfAttn ? 1 : 0);
      }
      gemm(yb, wbo, ab + 3 * CC, xf, hf, nullptr, MQ, CC, CC, 0);
      // mlp: h2 = h + W2 @ gelu(W1 @ ln(h))
      lnorm(hf, ln + 2 * CC, ln + 3 * CC, nullptr, t0b, MQ);
      gemm(t0b, wb1, b1, nullptr, nullptr, m1b, MQ, FF, CC, 1);
      gemm(m1b, wb2, b2, hf, h2f, nullptr, MQ, CC, FF, 0);
      // block-final layernorm
      if (selfAttn) {
        lnorm(h2f, ln + 4 * CC, ln + 5 * CC, xf, xb, MQ);   // x = ln(h2)
      } else {
        lnorm(h2f, ln + 4 * CC, ln + 5 * CC, blf, nullptr, MQ);
        k_add_inplace<<<(MQ * CC + 255) / 256, 256, 0, stream>>>(xf, blf, MQ * CC);
      }
    }
  }
  // final layernorm -> output
  lnorm(xf, lnf_g, lnf_b, out, nullptr, MQ);
}
